// VLAD_1425929142291
// MI455X (gfx1250) — matrix operand order repaired and checked by reading, not compiled, not run on hardware
//
#include <hip/hip_runtime.h>

// ---------------------------------------------------------------------------
// NetVLAD fused kernel for MI455X (gfx1250, wave32, WMMA 16x16x32 f16).
// GEMM1 (logits) + softmax + GEMM2 (assignment-weighted sum) fused so the
// 126MB feature map is read exactly once (~5.5us of HBM at 23.3 TB/s);
// both GEMMs run on v_wmma_f32_16x16x32_f16, with software-pipelined LDS
// fragment loads so ds_load_b128 overlaps the matrix pipe.
// ---------------------------------------------------------------------------

typedef _Float16 v16h __attribute__((ext_vector_type(16)));
typedef _Float16 v8h  __attribute__((ext_vector_type(8)));
typedef _Float16 v4h  __attribute__((ext_vector_type(4)));
typedef float    v8f  __attribute__((ext_vector_type(8)));

constexpr int BATCH  = 32;
constexpr int C      = 240;          // channels
constexpr int CP     = 256;          // padded channels (col 240 = ones column)
constexpr int K      = 64;           // clusters
constexpr int NPIX   = 4096;         // 64*64 spatial positions
constexpr int NT     = 128;          // positions per tile
constexpr int SPLITS = 8;            // workgroups per batch
constexpr int ITERS  = 4;            // n-tiles per workgroup (8*4*128 = 4096)

// LDS strides (halves), padded for bank-conflict avoidance, 16B-aligned rows.
constexpr int XNS = 264;             // xn[n][c]  (n-major, GEMM1 B operand)
constexpr int XCS = 136;             // xc[c][n]  (c-major, GEMM2 B operand)
constexpr int WS  = 264;             // w [k][c]  (GEMM1 A operand)
constexpr int MS  = 136;             // m [k][n]  (GEMM2 A operand)
constexpr int XN_H = NT * XNS;       // 33792 halves
constexpr int XC_H = CP * XCS;       // 34816
constexpr int W_H  = K  * WS;        //  16896
constexpr int M_H  = K  * MS;        //   8704
constexpr size_t SMEM1 = (size_t)(XN_H + XC_H + W_H + M_H) * 2 + K * 4; // ~188.7 KB

__device__ inline v8f vzero8() {
  v8f z;
  #pragma unroll
  for (int i = 0; i < 8; ++i) z[i] = 0.f;
  return z;
}

// A-matrix fragment (16x32 f16): lane row = lane%16,
// halves e=0..7 -> K = kb + o + e, e=8..15 -> K = kb + o + 16 + (e-8),
// o = (lane>=16) ? 8 : 0.   Two 16B LDS loads.
__device__ inline v16h frag_a(const _Float16* row, int kb, int hi) {
  const int o = kb + (hi ? 8 : 0);
  v8h lo = *(const v8h*)(row + o);
  v8h hh = *(const v8h*)(row + o + 16);
  v16h r;
  #pragma unroll
  for (int i = 0; i < 8; ++i) { r[i] = lo[i]; r[i + 8] = hh[i]; }
  return r;
}

// B-matrix fragment (32x16 f16): lane col = lane%16,
// halves e=0..7 -> K = kb + o + e, e=8..15 -> K = kb + o + 16 + (e-8),
// o = (lane>=16) ? 8 : 0.  Two 16B LDS loads.
__device__ inline v16h frag_b(const _Float16* row, int kb, int hi) {
  const int o = kb + (hi ? 8 : 0);
  v8h lo = *(const v8h*)(row + o);
  v8h hh = *(const v8h*)(row + o + 16);
  v16h r;
  #pragma unroll
  for (int i = 0; i < 8; ++i) { r[i] = lo[i]; r[i + 8] = hh[i]; }
  return r;
}

__device__ inline void atomicAddF32(float* p, float v) {
  __hip_atomic_fetch_add(p, v, __ATOMIC_RELAXED, __HIP_MEMORY_SCOPE_AGENT);
}

__global__ __launch_bounds__(256, 1)
void vlad_gemm_kernel(const float* __restrict__ fm, const float* __restrict__ mw,
                      const float* __restrict__ mb, float* __restrict__ out,
                      float* __restrict__ a_acc) {
  extern __shared__ char smem[];
  _Float16* xn = (_Float16*)smem;      // [NT][XNS]
  _Float16* xc = xn + XN_H;            // [CP][XCS]
  _Float16* wl = xc + XC_H;            // [K ][WS ]
  _Float16* ml = wl + W_H;             // [K ][MS ]
  float*  bias = (float*)(ml + M_H);   // [K]

  const int tid  = threadIdx.x;
  const int wave = tid >> 5, lane = tid & 31;
  const int l16  = lane & 15, hi = lane >> 4;
  const int b    = blockIdx.x / SPLITS;
  const int split= blockIdx.x % SPLITS;

  // ---- one-time setup: weights -> f16 LDS (zero-padded), bias, pad regions
  for (int i = tid; i < K * CP; i += 256) {
    const int k = i >> 8, c = i & 255;
    wl[k * WS + c] = (c < C) ? (_Float16)mw[k * C + c] : (_Float16)0.f;
  }
  if (tid < K) bias[tid] = mb[tid];
  for (int i = tid; i < 16 * NT; i += 256) {       // xc rows 240..255: ones col + zeros
    const int c = C + (i >> 7), n = i & (NT - 1);
    xc[c * XCS + n] = (c == C) ? (_Float16)1.f : (_Float16)0.f;
  }
  for (int i = tid; i < NT * 16; i += 256) {       // xn cols 240..255: zeros
    const int n = i >> 4, c = C + (i & 15);
    xn[n * XNS + c] = (_Float16)0.f;
  }

  // GEMM2 register accumulators: wave -> (kt2, 8 c-tiles), persist across ITERS
  const int kt2 = wave >> 1, ctb = (wave & 1) * 8;
  v8f acc2[8];
  #pragma unroll
  for (int j = 0; j < 8; ++j) acc2[j] = vzero8();

  for (int it = 0; it < ITERS; ++it) {
    const int nbase = (split * ITERS + it) * NT;
    __syncthreads();                               // previous tile fully consumed

    // ---- load x tile: lane-per-channel, f32 -> f16, dual LDS layouts
    if (tid < C) {
      const int c = tid;
      const float4* src = (const float4*)(fm + (size_t)(b * C + c) * NPIX + nbase);
      #pragma unroll 4
      for (int n4 = 0; n4 < NT / 4; ++n4) {
        const float4 v = src[n4];
        const _Float16 h0 = (_Float16)v.x, h1 = (_Float16)v.y;
        const _Float16 h2 = (_Float16)v.z, h3 = (_Float16)v.w;
        v4h p; p[0] = h0; p[1] = h1; p[2] = h2; p[3] = h3;
        *(v4h*)(xc + c * XCS + n4 * 4) = p;        // c-major, one b64
        const int n = n4 * 4;
        xn[(n + 0) * XNS + c] = h0;                // n-major scatter (b16 x4)
        xn[(n + 1) * XNS + c] = h1;
        xn[(n + 2) * XNS + c] = h2;
        xn[(n + 3) * XNS + c] = h3;
      }
    }
    __syncthreads();

    // ---- GEMM1: D[k,n] = w[k,c] * x[n,c]^T for this wave's n-strip.
    // Two-stage pipeline: stage cc+1 fragments load while stage cc WMMAs run.
    const int nt = wave;
    v8f acc1[4];
    #pragma unroll
    for (int kt = 0; kt < 4; ++kt) acc1[kt] = vzero8();
    const _Float16* xrow = xn + (nt * 16 + l16) * XNS;
    const _Float16* wrow = wl + l16 * WS;

    v16h aF[4], bF;
    bF = frag_b(xrow, 0, hi);
    #pragma unroll
    for (int kt = 0; kt < 4; ++kt) aF[kt] = frag_a(wrow + kt * 16 * WS, 0, hi);

    #pragma unroll
    for (int cc = 0; cc < CP / 32; ++cc) {
      v16h aN[4], bN;
      if (cc + 1 < CP / 32) {
        bN = frag_b(xrow, (cc + 1) * 32, hi);
        #pragma unroll
        for (int kt = 0; kt < 4; ++kt)
          aN[kt] = frag_a(wrow + kt * 16 * WS, (cc + 1) * 32, hi);
      }
      #pragma unroll
      for (int kt = 0; kt < 4; ++kt)
        acc1[kt] = __builtin_amdgcn_wmma_f32_16x16x32_f16(
            false, aF[kt], false, bF, (short)0, acc1[kt], false, false);
      if (cc + 1 < CP / 32) {
        bF = bN;
        #pragma unroll
        for (int kt = 0; kt < 4; ++kt) aF[kt] = aN[kt];
      }
    }

    // ---- bias + softmax over K=64 for position n = nt*16+l16.
    // This lane holds k = kt*16 + r + hi*8 (32 values); partner lane^16 has the rest.
    float v[32];
    float mx = -3.0e38f;
    #pragma unroll
    for (int kt = 0; kt < 4; ++kt) {
      #pragma unroll
      for (int r = 0; r < 8; ++r) {
        const float t = acc1[kt][r] + bias[kt * 16 + r + hi * 8];
        v[kt * 8 + r] = t;
        mx = fmaxf(mx, t);
      }
    }
    mx = fmaxf(mx, __shfl_xor(mx, 16, 32));
    float sum = 0.f;
    #pragma unroll
    for (int i = 0; i < 32; ++i) { const float e = __expf(v[i] - mx); v[i] = e; sum += e; }
    sum += __shfl_xor(sum, 16, 32);
    const float inv = 1.0f / sum;
    #pragma unroll
    for (int kt = 0; kt < 4; ++kt) {
      #pragma unroll
      for (int r = 0; r < 8; ++r)
        ml[(kt * 16 + r + hi * 8) * MS + nt * 16 + l16] =
            (_Float16)(v[kt * 8 + r] * inv);
    }

    __syncthreads();

    // ---- GEMM2: D[k,c] += m^T[k,n] * x[n,c]  (c=240 column == a[k], ones trick)
    // Same two-stage pipeline over the 4 n-chunks.
    const _Float16* mrow = ml + (kt2 * 16 + l16) * MS;
    const _Float16* xcb  = xc + l16 * XCS;

    v16h a2F, b2F[8];
    a2F = frag_a(mrow, 0, hi);
    #pragma unroll
    for (int j = 0; j < 8; ++j)
      b2F[j] = frag_b(xcb + (ctb + j) * 16 * XCS, 0, hi);

    #pragma unroll
    for (int nn = 0; nn < NT / 32; ++nn) {
      v16h a2N, b2N[8];
      if (nn + 1 < NT / 32) {
        a2N = frag_a(mrow, (nn + 1) * 32, hi);
        #pragma unroll
        for (int j = 0; j < 8; ++j)
          b2N[j] = frag_b(xcb + (ctb + j) * 16 * XCS, (nn + 1) * 32, hi);
      }
      #pragma unroll
      for (int j = 0; j < 8; ++j)
        acc2[j] = __builtin_amdgcn_wmma_f32_16x16x32_f16(
            false, a2F, false, b2F[j], (short)0, acc2[j], false, false);
      if (nn + 1 < NT / 32) {
        a2F = a2N;
        #pragma unroll
        for (int j = 0; j < 8; ++j) b2F[j] = b2N[j];
      }
    }
  }

  // ---- flush: atomics into d_out (= raw "weighted" accumulator) and a[b][k]
  #pragma unroll
  for (int j = 0; j < 8; ++j) {
    const int c = (ctb + j) * 16 + l16;
    #pragma unroll
    for (int r = 0; r < 8; ++r) {
      const int k = kt2 * 16 + r + hi * 8;
      const float val = acc2[j][r];
      if (c < C)       atomicAddF32(out + (size_t)b * (K * C) + k * C + c, val);
      else if (c == C) atomicAddF32(a_acc + b * K + k, val);
      // c > 240: exact zeros, skip
    }
  }
}

// ---------------------------------------------------------------------------
// Finalize: desc = a*clusters - weighted; intra L2-norm over C; global L2-norm.
// One workgroup per batch; LDS holds the 64x240 descriptor (61.4 KB).
// ---------------------------------------------------------------------------
__global__ __launch_bounds__(256, 1)
void vlad_finalize_kernel(float* __restrict__ out, const float* __restrict__ a_acc,
                          const float* __restrict__ clusters) {
  __shared__ float buf[K * C];
  __shared__ float rowsq[K];
  __shared__ float gscale;
  const int b = blockIdx.x, tid = threadIdx.x;
  const int wave = tid >> 5, lane = tid & 31;
  float* obase = out + (size_t)b * (K * C);

  for (int k = wave; k < K; k += 8) {
    const float a = a_acc[b * K + k];
    float val[8];
    float ssq = 0.f;
    #pragma unroll
    for (int j = 0; j < 8; ++j) {
      const int c = lane + 32 * j;
      float t = 0.f;
      if (c < C) t = a * clusters[k * C + c] - obase[k * C + c];
      val[j] = t;
      ssq += t * t;
    }
    #pragma unroll
    for (int off = 16; off > 0; off >>= 1) ssq += __shfl_xor(ssq, off, 32);
    const float s = 1.0f / fmaxf(sqrtf(ssq), 1e-12f);
    #pragma unroll
    for (int j = 0; j < 8; ++j) {
      const int c = lane + 32 * j;
      if (c < C) buf[k * C + c] = val[j] * s;
    }
    if (lane == 0) rowsq[k] = ssq * s * s;
  }
  __syncthreads();
  if (tid == 0) {
    float t = 0.f;
    for (int i = 0; i < K; ++i) t += rowsq[i];
    gscale = 1.0f / fmaxf(sqrtf(t), 1e-12f);
  }
  __syncthreads();
  const float g = gscale;
  for (int i = tid; i < K * C; i += 256) obase[i] = buf[i] * g;
}

extern "C" void kernel_launch(void* const* d_in, const int* in_sizes, int n_in,
                              void* d_out, int out_size, void* d_ws, size_t ws_size,
                              hipStream_t stream) {
  const float* fm = (const float*)d_in[0];   // feature_map [32,240,64,64]
  const float* mw = (const float*)d_in[1];   // mem_w [64,240]
  const float* mb = (const float*)d_in[2];   // mem_b [64]
  const float* cl = (const float*)d_in[3];   // clusters [1,64,240]
  float* out   = (float*)d_out;              // [32, 64*240]
  float* a_acc = (float*)d_ws;               // [32,64] softmax mass (8 KB)

  (void)hipMemsetAsync(out, 0, (size_t)BATCH * K * C * sizeof(float), stream);
  (void)hipMemsetAsync(a_acc, 0, (size_t)BATCH * K * sizeof(float), stream);

  vlad_gemm_kernel<<<dim3(BATCH * SPLITS), dim3(256), SMEM1, stream>>>(
      fm, mw, mb, out, a_acc);
  vlad_finalize_kernel<<<dim3(BATCH), dim3(256), 0, stream>>>(out, a_acc, cl);
}
